// PerTokenFFN_14998025798123
// MI455X (gfx1250) — matrix operand order repaired and checked by reading, compiled, not run on hardware
//
#include <hip/hip_runtime.h>
#include <math.h>

// ---------------------------------------------------------------------------
// Per-token FFN, fused:  y = GELU(x @ W1[t] + b1[t]) @ W2[t] + b2[t]
// B=2048, T=32, D=512, H=2048.
// Pass 1: convert x / W1^T / W2^T to bf16 in d_ws (coalesced tiled transpose).
// Pass 2: fused WMMA kernel; LDS staged by the Tensor Data Mover (TDM) with
//         double buffering + s_wait_tensorcnt, compute = v_wmma_f32_16x16x32_bf16.
// ---------------------------------------------------------------------------

typedef __attribute__((ext_vector_type(16))) __bf16        v16bf;
typedef __attribute__((ext_vector_type(8)))  float         v8f;
typedef __attribute__((ext_vector_type(4)))  unsigned int  v4u;
typedef __attribute__((ext_vector_type(8)))  int           v8i;
typedef __attribute__((ext_vector_type(4)))  int           v4i;

static constexpr int T_  = 32;
static constexpr int D_  = 512;
static constexpr int H_  = 2048;
static constexpr int BM  = 64;        // rows of x per workgroup
static constexpr int HC  = 32;        // H-chunk per iteration
static constexpr int NC  = H_ / HC;   // 64 chunks
static constexpr int XP  = D_ + 8;    // padded K stride (1040B, 16B multiple)
static constexpr int KP  = HC + 8;    // padded K stride (80B, 16B multiple)

#if defined(__HIP_DEVICE_COMPILE__) && defined(__has_builtin)
#  if __has_builtin(__builtin_amdgcn_tensor_load_to_lds) && \
      __has_builtin(__builtin_amdgcn_s_wait_tensorcnt)
#    define USE_TDM 1
#  endif
#endif
#ifndef USE_TDM
#  define USE_TDM 0
#endif

__device__ __forceinline__ unsigned short f2bf(float f) {
    unsigned int u = __float_as_uint(f);
    u += 0x7FFFu + ((u >> 16) & 1u);           // round-to-nearest-even
    return (unsigned short)(u >> 16);
}

__device__ __forceinline__ float gelu_erf(float v) {
    return 0.5f * v * (1.0f + erff(v * 0.70710678118654752f));
}

__device__ __forceinline__ v16bf load_frag(const unsigned short* p0,
                                           const unsigned short* p1) {
    union { v16bf v; v4u u[2]; } r;
    r.u[0] = *reinterpret_cast<const v4u*>(p0);
    r.u[1] = *reinterpret_cast<const v4u*>(p1);
    return r.v;
}

#if USE_TDM
// 2D tile load via Tensor Data Mover.  Descriptor per CDNA5 ISA §8:
//   group0: count=1 | lds_addr | global_addr | type=2
//   group1: data_size=2B, pad_enable, pad_interval/amount, tensor/tile dims,
//           tensor_dim0_stride (elements).  Tensor dims == tile dims (no OOB).
// This toolchain exposes the 6-arg builtin:
//   (u32x4 g0, i32x8 g1, i32x4 g2, i32x4 g3, i32x8 extra, i32 cpol)
// 2D tensors leave groups 2/3 (and the extra group) zero.
__device__ __forceinline__ void tdm_load_2d(unsigned lds_off, const void* gp,
                                            unsigned tile_k, unsigned tile_rows,
                                            unsigned long long stride_elems,
                                            unsigned pad_interval,
                                            unsigned pad_amount) {
    unsigned long long ga = (unsigned long long)(uintptr_t)gp;
    v4u g0 = { 1u,                                   // count=1 (valid, user)
               lds_off,                              // lds_addr
               (unsigned)(ga & 0xFFFFFFFFu),         // global_addr[31:0]
               (unsigned)((ga >> 32) & 0x1FFFFFFu) | (2u << 30) };  // type=2
    v8i g1;
    g1[0] = (int)((1u << 16)                         // data_size = 2 bytes
                | (1u << 20)                         // pad_enable
                | (pad_interval << 22)
                | (pad_amount << 25));
    g1[1] = (int)((tile_k & 0xFFFFu) << 16);                         // tdim0 lo
    g1[2] = (int)((tile_k >> 16) | ((tile_rows & 0xFFFFu) << 16));   // tdim0 hi | tdim1 lo
    g1[3] = (int)((tile_rows >> 16) | ((tile_k & 0xFFFFu) << 16));   // tdim1 hi | tile_dim0
    g1[4] = (int)(tile_rows & 0xFFFFu);                              // tile_dim1 (tile_dim2=0)
    g1[5] = (int)(stride_elems & 0xFFFFFFFFull);                     // dim0_stride lo
    g1[6] = (int)((stride_elems >> 32) & 0xFFFFull);                 // dim0_stride hi
    g1[7] = 0;
    v4i z4 = { 0, 0, 0, 0 };
    v8i z8 = { 0, 0, 0, 0, 0, 0, 0, 0 };
    __builtin_amdgcn_tensor_load_to_lds(g0, g1, z4, z4, z8, 0);
}
#endif

// ---------------------------------------------------------------------------
// Pre-pass kernels: f32 -> bf16 (x) and tiled transpose f32 -> bf16 (weights)
// ---------------------------------------------------------------------------
__global__ void convert_bf16(const float* __restrict__ in,
                             unsigned short* __restrict__ out, long long n) {
    for (long long i = (long long)blockIdx.x * blockDim.x + threadIdx.x;
         i < n; i += (long long)gridDim.x * blockDim.x)
        out[i] = f2bf(in[i]);
}

// W: [T][R][C] f32  ->  WT: [T][C][R] bf16   (R, C multiples of 32)
__global__ __launch_bounds__(256)
void transpose_bf16(const float* __restrict__ W, unsigned short* __restrict__ WT,
                    int R, int C) {
    __shared__ float tile[32][33];
    const int t  = blockIdx.z;
    const int c0 = blockIdx.x * 32, r0 = blockIdx.y * 32;
    const int tx = threadIdx.x, ty = threadIdx.y;     // block (32, 8)
    #pragma unroll
    for (int j = 0; j < 4; ++j)
        tile[ty + 8 * j][tx] =
            W[((size_t)t * R + (r0 + ty + 8 * j)) * C + c0 + tx];
    __syncthreads();
    #pragma unroll
    for (int j = 0; j < 4; ++j)
        WT[((size_t)t * C + (c0 + ty + 8 * j)) * R + r0 + tx] =
            f2bf(tile[tx][ty + 8 * j]);
}

// ---------------------------------------------------------------------------
// Main fused kernel: bf16 inputs from d_ws, TDM-staged LDS, WMMA compute.
// ---------------------------------------------------------------------------
__global__ __launch_bounds__(256)
void ffn_wmma_tdm(const unsigned short* __restrict__ xbf,   // [B][T][D]
                  const unsigned short* __restrict__ W1T,   // [T][H][D]
                  const float* __restrict__ b1,
                  const unsigned short* __restrict__ W2T,   // [T][D][H]
                  const float* __restrict__ b2,
                  float* __restrict__ y) {
    const int t    = blockIdx.y;
    const int b0   = blockIdx.x * BM;
    const int tid  = threadIdx.x;
    const int lane = tid & 31;
    const int wave = tid >> 5;
    const int half = lane >> 4;
    const int l16  = lane & 15;

    __shared__ unsigned short sX  [BM][XP];      // x tile       [m][k0..511]
    __shared__ unsigned short sW1c[2][HC][XP];   // W1^T chunk   [n][k0..511]
    __shared__ unsigned short sW2c[2][D_][KP];   // W2^T chunk   [n][k0..31]
    __shared__ unsigned short sH  [BM][KP];      // GELU(h)      [m][k0..31]

    v8f acc[4][4] = {};

#if USE_TDM
    if (wave == 0) {
        // x tile: BM rows of D_ bf16, row stride T_*D_; 1024B rows -> pad code 7/3
        tdm_load_2d((unsigned)(uintptr_t)&sX[0][0],
                    &xbf[((size_t)b0 * T_ + t) * D_],
                    D_, BM, (unsigned long long)T_ * D_, 7u, 3u);
        tdm_load_2d((unsigned)(uintptr_t)&sW1c[0][0][0],
                    &W1T[(size_t)t * H_ * D_],
                    D_, HC, (unsigned long long)D_, 7u, 3u);
        // W2^T chunk: D_ rows of HC bf16 (64B rows -> pad code 3/3), stride H_
        tdm_load_2d((unsigned)(uintptr_t)&sW2c[0][0][0],
                    &W2T[(size_t)t * D_ * H_],
                    HC, D_, (unsigned long long)H_, 3u, 3u);
    }
#else
    for (int i = tid; i < BM * D_; i += 256) {
        const int m = i >> 9, k = i & (D_ - 1);
        sX[m][k] = xbf[((size_t)(b0 + m) * T_ + t) * D_ + k];
    }
#endif

    const int mT1 = wave >> 1;     // GEMM1 tile of this wave
    const int nT1 = wave & 1;

    for (int ci = 0; ci < NC; ++ci) {
        const int hc  = ci * HC;
        const int buf = ci & 1;
        __syncthreads();                       // prev GEMM2 done with buf^1
#if USE_TDM
        if (wave == 0) {
            if (ci + 1 < NC) {
                const int nb = buf ^ 1, nhc = hc + HC;
                tdm_load_2d((unsigned)(uintptr_t)&sW1c[nb][0][0],
                            &W1T[((size_t)t * H_ + nhc) * D_],
                            D_, HC, (unsigned long long)D_, 7u, 3u);
                tdm_load_2d((unsigned)(uintptr_t)&sW2c[nb][0][0],
                            &W2T[(size_t)t * D_ * H_ + nhc],
                            HC, D_, (unsigned long long)H_, 3u, 3u);
                __builtin_amdgcn_s_wait_tensorcnt(2);   // chunk ci resident
            } else {
                __builtin_amdgcn_s_wait_tensorcnt(0);
            }
        }
#else
        for (int i = tid; i < HC * D_; i += 256) {
            const int n = i >> 9, k = i & (D_ - 1);
            sW1c[buf][n][k] = W1T[((size_t)t * H_ + hc + n) * D_ + k];
        }
        for (int i = tid; i < HC * D_; i += 256) {
            const int n = i >> 5, k = i & (HC - 1);
            sW2c[buf][n][k] = W2T[((size_t)t * D_ + n) * H_ + hc + k];
        }
#endif
        __syncthreads();                       // chunk ci visible to all waves

        // ---- GEMM1: h = x @ W1c  (one 16x16 tile per wave, K = 512) -----
        {
            v8f c = {};
            const int mrow = mT1 * 16 + l16;
            const int nrow = nT1 * 16 + l16;
            #pragma unroll 4
            for (int kk = 0; kk < D_; kk += 32) {
                v16bf a = load_frag(&sX[mrow][kk + half * 8],
                                    &sX[mrow][kk + 16 + half * 8]);
                v16bf b = load_frag(&sW1c[buf][nrow][kk + half * 8],
                                    &sW1c[buf][nrow][kk + 16 + half * 8]);
                c = __builtin_amdgcn_wmma_f32_16x16x32_bf16(
                        false, a, false, b, (short)0, c, false, false);
            }
            const int   ncol = nT1 * 16 + l16;
            const float bias = b1[t * H_ + hc + ncol];
            #pragma unroll
            for (int r = 0; r < 8; ++r) {
                const float v = gelu_erf(c[r] + bias);
                sH[mT1 * 16 + r + 8 * half][ncol] = f2bf(v);
            }
        }
        __syncthreads();                       // sH ready

        // ---- GEMM2: y += h @ W2c  (wave owns 4 mTiles x 4 nTiles) -------
        {
            v16bf aF[4];
            #pragma unroll
            for (int mT = 0; mT < 4; ++mT) {
                const int mrow = mT * 16 + l16;
                aF[mT] = load_frag(&sH[mrow][half * 8],
                                   &sH[mrow][16 + half * 8]);
            }
            #pragma unroll
            for (int nn = 0; nn < 4; ++nn) {
                const int col = (wave * 4 + nn) * 16 + l16;
                v16bf bF = load_frag(&sW2c[buf][col][half * 8],
                                     &sW2c[buf][col][16 + half * 8]);
                #pragma unroll
                for (int mT = 0; mT < 4; ++mT) {
                    acc[mT][nn] = __builtin_amdgcn_wmma_f32_16x16x32_bf16(
                        false, aF[mT], false, bF, (short)0, acc[mT][nn],
                        false, false);
                }
            }
        }
    }

    // ---- epilogue: + b2, store y f32 -----------------------------------
    #pragma unroll
    for (int nn = 0; nn < 4; ++nn) {
        const int n = (wave * 4 + nn) * 16 + l16;
        const float bias = b2[t * D_ + n];
        #pragma unroll
        for (int mT = 0; mT < 4; ++mT) {
            #pragma unroll
            for (int r = 0; r < 8; ++r) {
                const int row = b0 + mT * 16 + r + 8 * half;
                y[((size_t)row * T_ + t) * D_ + n] = acc[mT][nn][r] + bias;
            }
        }
    }
}

// ---------------------------------------------------------------------------
// Fallback: fully fused f32-input kernel (no workspace needed).
// ---------------------------------------------------------------------------
__global__ __launch_bounds__(256)
void ffn_fused_wmma(const float* __restrict__ x,
                    const float* __restrict__ W1,
                    const float* __restrict__ b1,
                    const float* __restrict__ W2,
                    const float* __restrict__ b2,
                    float* __restrict__ y) {
    const int t    = blockIdx.y;
    const int b0   = blockIdx.x * BM;
    const int tid  = threadIdx.x;
    const int lane = tid & 31;
    const int wave = tid >> 5;
    const int half = lane >> 4;
    const int l16  = lane & 15;

    __shared__ unsigned short sX  [BM][XP];
    __shared__ unsigned short sW1c[HC][XP];
    __shared__ unsigned short sW2c[D_][KP];
    __shared__ unsigned short sH  [BM][KP];

    for (int i = tid; i < BM * D_; i += 256) {
        const int m = i >> 9, k = i & (D_ - 1);
        sX[m][k] = f2bf(x[((size_t)(b0 + m) * T_ + t) * D_ + k]);
    }
    v8f acc[4][4] = {};
    __syncthreads();

    const int mT1 = wave >> 1, nT1 = wave & 1;

    for (int hc = 0; hc < H_; hc += HC) {
        for (int i = tid; i < HC * D_; i += 256) {
            const int n = i & (HC - 1), k = i >> 5;
            sW1c[n][k] = f2bf(W1[((size_t)t * D_ + k) * H_ + hc + n]);
        }
        for (int i = tid; i < HC * D_; i += 256) {
            const int n = i & (D_ - 1), k = i >> 9;
            sW2c[n][k] = f2bf(W2[((size_t)t * H_ + hc + k) * D_ + n]);
        }
        __syncthreads();
        {
            v8f c = {};
            const int mrow = mT1 * 16 + l16;
            const int nrow = nT1 * 16 + l16;
            #pragma unroll 4
            for (int kk = 0; kk < D_; kk += 32) {
                v16bf a = load_frag(&sX[mrow][kk + half * 8],
                                    &sX[mrow][kk + 16 + half * 8]);
                v16bf b = load_frag(&sW1c[nrow][kk + half * 8],
                                    &sW1c[nrow][kk + 16 + half * 8]);
                c = __builtin_amdgcn_wmma_f32_16x16x32_bf16(
                        false, a, false, b, (short)0, c, false, false);
            }
            const int   ncol = nT1 * 16 + l16;
            const float bias = b1[t * H_ + hc + ncol];
            #pragma unroll
            for (int r = 0; r < 8; ++r)
                sH[mT1 * 16 + r + 8 * half][ncol] = f2bf(gelu_erf(c[r] + bias));
        }
        __syncthreads();
        {
            v16bf aF[4];
            #pragma unroll
            for (int mT = 0; mT < 4; ++mT) {
                const int mrow = mT * 16 + l16;
                aF[mT] = load_frag(&sH[mrow][half * 8], &sH[mrow][16 + half * 8]);
            }
            #pragma unroll
            for (int nn = 0; nn < 4; ++nn) {
                const int col = (wave * 4 + nn) * 16 + l16;
                v16bf bF = load_frag(&sW2c[col][half * 8],
                                     &sW2c[col][16 + half * 8]);
                #pragma unroll
                for (int mT = 0; mT < 4; ++mT)
                    acc[mT][nn] = __builtin_amdgcn_wmma_f32_16x16x32_bf16(
                        false, aF[mT], false, bF, (short)0, acc[mT][nn],
                        false, false);
            }
        }
        __syncthreads();
    }
    #pragma unroll
    for (int nn = 0; nn < 4; ++nn) {
        const int n = (wave * 4 + nn) * 16 + l16;
        const float bias = b2[t * D_ + n];
        #pragma unroll
        for (int mT = 0; mT < 4; ++mT)
            #pragma unroll
            for (int r = 0; r < 8; ++r) {
                const int row = b0 + mT * 16 + r + 8 * half;
                y[((size_t)row * T_ + t) * D_ + n] = acc[mT][nn][r] + bias;
            }
    }
}

// ---------------------------------------------------------------------------
extern "C" void kernel_launch(void* const* d_in, const int* in_sizes, int n_in,
                              void* d_out, int out_size, void* d_ws, size_t ws_size,
                              hipStream_t stream) {
    const float* x  = (const float*)d_in[0];
    const float* W1 = (const float*)d_in[1];
    const float* b1 = (const float*)d_in[2];
    const float* W2 = (const float*)d_in[3];
    const float* b2 = (const float*)d_in[4];
    float* y = (float*)d_out;

    const int B = in_sizes[0] / (T_ * D_);             // 2048
    const long long xN = (long long)B * T_ * D_;       // 33.5M
    const long long wN = (long long)T_ * D_ * H_;      // 33.5M
    const size_t need = (size_t)(xN + 2 * wN) * sizeof(unsigned short);

    dim3 grid(B / BM, T_);
    dim3 block(256);

    if (ws_size >= need) {
        unsigned short* xbf = (unsigned short*)d_ws;
        unsigned short* W1T = xbf + xN;
        unsigned short* W2T = W1T + wN;

        convert_bf16<<<4096, 256, 0, stream>>>(x, xbf, xN);
        // W1: [T][D][H] -> W1T: [T][H][D]
        transpose_bf16<<<dim3(H_ / 32, D_ / 32, T_), dim3(32, 8), 0, stream>>>(
            W1, W1T, D_, H_);
        // W2: [T][H][D] -> W2T: [T][D][H]
        transpose_bf16<<<dim3(D_ / 32, H_ / 32, T_), dim3(32, 8), 0, stream>>>(
            W2, W2T, H_, D_);

        ffn_wmma_tdm<<<grid, block, 0, stream>>>(xbf, W1T, b1, W2T, b2, y);
    } else {
        ffn_fused_wmma<<<grid, block, 0, stream>>>(x, W1, b1, W2, b2, y);
    }
}
